// BaseMoEModel_52888227283713
// MI455X (gfx1250) — compile-verified, matrix operand order checked by reading, not run on hardware
//
#include <hip/hip_runtime.h>
#include <math.h>

// ---------------- model dims ----------------
#define Vn   32000
#define Dn   512
#define Hn   8
#define DHn  64
#define Ln   12
#define En   8
#define FFn  2048
#define Bq   2
#define Sq   1024
#define Tn   (Bq * Sq)
#define NEGV (-1e9f)

typedef __attribute__((ext_vector_type(16))) _Float16 v16h;
typedef __attribute__((ext_vector_type(8)))  float    v8f;

union FragU { v16h v; unsigned u[8]; };

// CDNA5 16-bit A-matrix 16x32 layout (ISA 7.12.2):
//  lanes 0-15 : row M=lane,  K pairs {0,2,4,6, 16,18,20,22}
//  lanes 16-31: row M=lane-16, same pairs shifted by +8.
// Per lane this is two contiguous 16-byte chunks -> two global_load_b128.
// B is staged pre-transposed ([N,K]) so the same loader serves both operands.
__device__ __forceinline__ v16h frag_load(const _Float16* rowp, int half) {
  FragU f;
  const int kb = half * 8;
#pragma unroll
  for (int j = 0; j < 8; ++j) {
    const int k = kb + (j & 3) * 2 + (j >> 2) * 16;
    f.u[j] = *reinterpret_cast<const unsigned*>(rowp + k);
  }
  return f.v;
}

// ---------------- generic WMMA GEMM ----------------
#define WF_GELU    1
#define WF_ACC     2
#define WF_CAUSAL  4
#define WF_SCATTER 8

struct GemmP {
  const _Float16* A;  long long sA;  int lda;
  const _Float16* Bw; long long sB;  int ldb; int nrowsB;
  float* Cf; _Float16* Cb;
  long long sC1, sC2; int zdiv; int ldcf; int ldcb;
  const float* bias;  long long sBias;
  const int* gather;  int gS;
  const int* cnt;
  const float* gate;
  int M, N, K;
  float alpha;
  int flags;
};

// 256 threads = 8 wave32; wave tile 16(M)x64(N); block macro-tile 64x128.
// K-loop: unroll-by-2 ping-pong double buffer (K/32 is even for every GEMM in
// this model), so next-step fragment loads overlap current WMMAs with zero
// register-rotation movs.
__global__ void gemm_f16_wmma(GemmP p) {
  const int lane = threadIdx.x & 31;
  const int wave = threadIdx.x >> 5;
  const int wm = wave & 3, wn = wave >> 2;
  const int z  = blockIdx.z;
  const int m0 = blockIdx.y * 64  + wm * 16;
  const int n0 = blockIdx.x * 128 + wn * 64;
  const int mvalid0 = p.cnt ? p.cnt[z] : p.M;
  const int mvalid  = (mvalid0 < p.M) ? mvalid0 : p.M;
  if (m0 >= mvalid) return;
  if (n0 >= p.N)    return;

  const long long cbase = (long long)(z / p.zdiv) * p.sC1
                        + (long long)(z % p.zdiv) * p.sC2;
  const int r = lane & 15, half = lane >> 4;

  // A fragment row (gathered for expert dispatch, clamped for ragged tiles)
  int am = m0 + r; am = (am < mvalid - 1) ? am : (mvalid - 1);
  const int arow = p.gather ? p.gather[(long long)z * p.gS + am] : am;
  const _Float16* Arow = p.A + (long long)z * p.sA + (long long)arow * p.lda;

  const _Float16* Brow[4];
#pragma unroll
  for (int t = 0; t < 4; ++t) {
    int bn = n0 + t * 16 + r;
    bn = (bn < p.nrowsB - 1) ? bn : (p.nrowsB - 1);
    Brow[t] = p.Bw + (long long)z * p.sB + (long long)bn * p.ldb;
  }

  v8f acc[4] = {};
  const bool fullmask = (p.flags & WF_CAUSAL) && (n0 > m0 + 15);
  if (!fullmask) {
    // set0 <- k = 0
    v16h a0 = frag_load(Arow, half);
    v16h b0[4];
#pragma unroll
    for (int t = 0; t < 4; ++t) b0[t] = frag_load(Brow[t], half);

    v16h a1; v16h b1[4];
    int k = 0;
    for (; k + 64 < p.K; k += 64) {
      if (k + 192 < p.K) {
        __builtin_prefetch(Arow + k + 192, 0, 1);      // global_prefetch_b8
        __builtin_prefetch(Brow[0] + k + 192, 0, 1);
      }
      // set1 <- k+32, consume set0
      a1 = frag_load(Arow + k + 32, half);
#pragma unroll
      for (int t = 0; t < 4; ++t) b1[t] = frag_load(Brow[t] + k + 32, half);
#pragma unroll
      for (int t = 0; t < 4; ++t)
        acc[t] = __builtin_amdgcn_wmma_f32_16x16x32_f16(
            false, a0, false, b0[t], (short)0, acc[t], false, false);
      // set0 <- k+64, consume set1
      a0 = frag_load(Arow + k + 64, half);
#pragma unroll
      for (int t = 0; t < 4; ++t) b0[t] = frag_load(Brow[t] + k + 64, half);
#pragma unroll
      for (int t = 0; t < 4; ++t)
        acc[t] = __builtin_amdgcn_wmma_f32_16x16x32_f16(
            false, a1, false, b1[t], (short)0, acc[t], false, false);
    }
    // tail: exactly two K-steps remain (K/32 even)
    a1 = frag_load(Arow + k + 32, half);
#pragma unroll
    for (int t = 0; t < 4; ++t) b1[t] = frag_load(Brow[t] + k + 32, half);
#pragma unroll
    for (int t = 0; t < 4; ++t)
      acc[t] = __builtin_amdgcn_wmma_f32_16x16x32_f16(
          false, a0, false, b0[t], (short)0, acc[t], false, false);
#pragma unroll
    for (int t = 0; t < 4; ++t)
      acc[t] = __builtin_amdgcn_wmma_f32_16x16x32_f16(
          false, a1, false, b1[t], (short)0, acc[t], false, false);
  }

  // C/D f32 layout: VGPR rr, lanes0-15 -> M=m0+rr, lanes16-31 -> M=m0+rr+8
#pragma unroll
  for (int t = 0; t < 4; ++t) {
    const int n = n0 + t * 16 + r;
    if (n >= p.N) continue;
#pragma unroll
    for (int rr = 0; rr < 8; ++rr) {
      const int m = m0 + rr + half * 8;
      if (m >= mvalid) continue;
      float v = acc[t][rr] * p.alpha;
      if ((p.flags & WF_CAUSAL) && (n > m)) v = NEGV;
      if (p.bias) v += p.bias[(long long)z * p.sBias + n];
      if (p.flags & WF_GELU) v = 0.5f * v * (1.f + erff(v * 0.70710678118f));
      int orow = m; float scale = 1.f;
      if (p.flags & WF_SCATTER) {
        orow  = p.gather[(long long)z * p.gS + m];
        scale = p.gate[orow];
      }
      if (p.Cf) {
        const long long idx = cbase + (long long)orow * p.ldcf + n;
        if (p.flags & WF_ACC) p.Cf[idx] += scale * v; else p.Cf[idx] = v;
      }
      if (p.Cb) {
        const long long idx = cbase + (long long)orow * p.ldcb + n;
        p.Cb[idx] = (_Float16)v;
      }
    }
  }
}

// ---------------- support kernels ----------------
__global__ void convT_kernel(const float* in, _Float16* out, int K, int Nin,
                             int Nout, long long sIn, long long sOut) {
  const long long total = (long long)Nout * K;
  const long long id = (long long)blockIdx.x * blockDim.x + threadIdx.x;
  if (id >= total) return;
  const int n = (int)(id / K), k = (int)(id % K);
  const float* src = in + (long long)blockIdx.z * sIn;
  const float v = (n < Nin) ? src[(long long)k * Nin + n] : 0.f;
  out[(long long)blockIdx.z * sOut + (long long)n * K + k] = (_Float16)v;
}

__global__ void embed_kernel(const int* ids, const float* tok, const float* pos,
                             float* x) {
  const long long id = (long long)blockIdx.x * blockDim.x + threadIdx.x;
  if (id >= (long long)Tn * Dn) return;
  const int t = (int)(id / Dn), d = (int)(id % Dn);
  const int s = t % Sq;
  x[id] = tok[(long long)ids[t] * Dn + d] + pos[(long long)s * Dn + d];
}

__global__ void rmsnorm_kernel(const float* x, const float* w, _Float16* out) {
  const int t = blockIdx.x;
  const float* xr = x + (long long)t * Dn;
  __shared__ float red[256];
  float s = 0.f;
  for (int i = threadIdx.x; i < Dn; i += 256) { const float v = xr[i]; s += v * v; }
  red[threadIdx.x] = s; __syncthreads();
  for (int k = 128; k > 0; k >>= 1) {
    if (threadIdx.x < k) red[threadIdx.x] += red[threadIdx.x + k];
    __syncthreads();
  }
  const float scale = rsqrtf(red[0] / Dn + 1e-6f);
  for (int i = threadIdx.x; i < Dn; i += 256)
    out[(long long)t * Dn + i] = (_Float16)(xr[i] * scale * w[i]);
}

__global__ void split_qkv_kernel(const _Float16* qkv, _Float16* q, _Float16* k,
                                 _Float16* vt) {
  const long long id = (long long)blockIdx.x * blockDim.x + threadIdx.x;
  if (id >= (long long)Tn * Dn) return;
  const int t = (int)(id / Dn), hd = (int)(id % Dn);
  const int h = hd / DHn, d = hd % DHn;
  const int b = t / Sq, s = t % Sq;
  const long long base = (long long)t * 3 * Dn + h * DHn + d;
  const long long bh = (long long)(b * Hn + h);
  q [(bh * Sq + s) * DHn + d] = qkv[base];
  k [(bh * Sq + s) * DHn + d] = qkv[base + Dn];
  vt[(bh * DHn + d) * Sq + s] = qkv[base + 2 * Dn];
}

__global__ void softmax_rows_kernel(const float* Sc, _Float16* P) {
  const int row = blockIdx.x, z = blockIdx.y;
  const float* src = Sc + ((long long)z * Sq + row) * Sq;
  _Float16* dst = P + ((long long)z * Sq + row) * Sq;
  __shared__ float red[256];
  float mx = -1e30f;
  for (int i = threadIdx.x; i < Sq; i += 256) mx = fmaxf(mx, src[i]);
  red[threadIdx.x] = mx; __syncthreads();
  for (int k = 128; k > 0; k >>= 1) {
    if (threadIdx.x < k) red[threadIdx.x] = fmaxf(red[threadIdx.x], red[threadIdx.x + k]);
    __syncthreads();
  }
  mx = red[0]; __syncthreads();
  float sum = 0.f;
  for (int i = threadIdx.x; i < Sq; i += 256) sum += expf(src[i] - mx);
  red[threadIdx.x] = sum; __syncthreads();
  for (int k = 128; k > 0; k >>= 1) {
    if (threadIdx.x < k) red[threadIdx.x] += red[threadIdx.x + k];
    __syncthreads();
  }
  const float inv = 1.f / red[0];
  for (int i = threadIdx.x; i < Sq; i += 256)
    dst[i] = (_Float16)(expf(src[i] - mx) * inv);
}

__global__ void zero32_kernel(unsigned* p, int n) {
  const int i = blockIdx.x * blockDim.x + threadIdx.x;
  if (i < n) p[i] = 0u;
}

__global__ void router_top1_kernel(const float* rl, float* top_w, int* cnt,
                                   int* lists, float* probSum) {
  const int t = blockIdx.x * blockDim.x + threadIdx.x;
  if (t >= Tn) return;
  const float* r = rl + (long long)t * En;
  float m = r[0]; int bi = 0;
  for (int e = 1; e < En; ++e) if (r[e] > m) { m = r[e]; bi = e; }
  float p[En], denom = 0.f;
  for (int e = 0; e < En; ++e) { p[e] = expf(r[e] - m); denom += p[e]; }
  const float inv = 1.f / denom;
  for (int e = 0; e < En; ++e) atomicAdd(&probSum[e], p[e] * inv);
  top_w[t] = p[bi] * inv;
  const int slot = atomicAdd(&cnt[bi], 1);
  lists[bi * Tn + slot] = t;
}

__global__ void aux_accum_kernel(const int* cnt, const float* probSum, float* aux) {
  if (threadIdx.x == 0 && blockIdx.x == 0) {
    float s = 0.f;
    for (int e = 0; e < En; ++e)
      s += ((float)cnt[e] / Tn) * (probSum[e] / Tn);
    aux[0] += (float)En * s;
  }
}

// ---------------- host ----------------
extern "C" void kernel_launch(void* const* d_in, const int* in_sizes, int n_in,
                              void* d_out, int out_size, void* d_ws, size_t ws_size,
                              hipStream_t stream) {
  (void)in_sizes; (void)n_in; (void)out_size; (void)ws_size;
  const int*   ids  = (const int*)  d_in[0];
  const float* tok  = (const float*)d_in[1];
  const float* pos  = (const float*)d_in[2];
  const float* Wqkv = (const float*)d_in[3];
  const float* bqkv = (const float*)d_in[4];
  const float* Wo   = (const float*)d_in[5];
  const float* bo   = (const float*)d_in[6];
  const float* rW   = (const float*)d_in[7];
  const float* rb   = (const float*)d_in[8];
  const float* eW1  = (const float*)d_in[9];
  const float* eb1  = (const float*)d_in[10];
  const float* eW2  = (const float*)d_in[11];
  const float* eb2  = (const float*)d_in[12];
  const float* n1w  = (const float*)d_in[13];
  const float* n2w  = (const float*)d_in[14];
  const float* now  = (const float*)d_in[15];
  const float* lmW  = (const float*)d_in[16];
  float* out = (float*)d_out;

  char* wsb = (char*)d_ws; size_t cur = 0;
  auto alloc = [&](size_t bytes) -> char* {
    char* p = wsb + cur; cur += (bytes + 255) & ~(size_t)255; return p;
  };

  // transposed f16 weights
  _Float16* wqkvT = (_Float16*)alloc((size_t)Ln * 3 * Dn * Dn * 2);
  _Float16* woT   = (_Float16*)alloc((size_t)Ln * Dn * Dn * 2);
  _Float16* rwT   = (_Float16*)alloc((size_t)Ln * 16 * Dn * 2);      // padded E->16
  _Float16* w1T   = (_Float16*)alloc((size_t)Ln * En * FFn * Dn * 2);
  _Float16* w2T   = (_Float16*)alloc((size_t)Ln * En * Dn * FFn * 2);
  _Float16* lmWT  = (_Float16*)alloc((size_t)Vn * Dn * 2);
  // activations
  float*    x     = (float*)   alloc((size_t)Tn * Dn * 4);
  _Float16* hb    = (_Float16*)alloc((size_t)Tn * Dn * 2);
  _Float16* qkvb  = (_Float16*)alloc((size_t)Tn * 3 * Dn * 2);
  _Float16* qb    = (_Float16*)alloc((size_t)Bq * Hn * Sq * DHn * 2);
  _Float16* kb    = (_Float16*)alloc((size_t)Bq * Hn * Sq * DHn * 2);
  _Float16* vt    = (_Float16*)alloc((size_t)Bq * Hn * DHn * Sq * 2);
  float*    scF   = (float*)   alloc((size_t)Bq * Hn * Sq * Sq * 4);
  _Float16* pb    = (_Float16*)alloc((size_t)Bq * Hn * Sq * Sq * 2);
  _Float16* ob    = (_Float16*)alloc((size_t)Tn * Dn * 2);
  _Float16* hid   = (_Float16*)alloc((size_t)En * Tn * FFn * 2);
  int*      cnt   = (int*)     alloc(64);   // 8 int + 8 float
  float*    probSum = (float*)(cnt + En);
  float*    top_w = (float*)   alloc((size_t)Tn * 4);
  int*      lists = (int*)     alloc((size_t)En * Tn * 4);

  float* aux_out  = out + (size_t)Tn * Vn;
  float* rlog_out = aux_out + 1;

  auto convT = [&](const float* in, _Float16* o, int K, int Nin, int Nout, int Z) {
    const long long tot = (long long)Nout * K;
    dim3 g((unsigned)((tot + 255) / 256), 1, (unsigned)Z);
    convT_kernel<<<g, 256, 0, stream>>>(in, o, K, Nin, Nout,
                                        (long long)K * Nin, (long long)Nout * K);
  };
  convT(Wqkv, wqkvT, Dn, 3 * Dn, 3 * Dn, Ln);
  convT(Wo,   woT,   Dn, Dn,     Dn,     Ln);
  convT(rW,   rwT,   Dn, En,     16,     Ln);
  convT(eW1,  w1T,   Dn, FFn,    FFn,    Ln * En);
  convT(eW2,  w2T,   FFn, Dn,    Dn,     Ln * En);
  convT(lmW,  lmWT,  Dn, Vn,     Vn,     1);

  embed_kernel<<<(Tn * Dn + 255) / 256, 256, 0, stream>>>(ids, tok, pos, x);
  zero32_kernel<<<1, 32, 0, stream>>>((unsigned*)aux_out, 1);

  auto gemm = [&](const _Float16* A, long long sA, int lda,
                  const _Float16* Bw, long long sB, int ldb, int nrowsB,
                  float* Cf, _Float16* Cb, long long sC1, long long sC2,
                  int zdiv, int ldcf, int ldcb,
                  const float* bias, long long sBias,
                  const int* gather, int gS, const int* cntp, const float* gate,
                  int M, int N, int K, float alpha, int flags, int Z) {
    GemmP p;
    p.A = A; p.sA = sA; p.lda = lda;
    p.Bw = Bw; p.sB = sB; p.ldb = ldb; p.nrowsB = nrowsB;
    p.Cf = Cf; p.Cb = Cb; p.sC1 = sC1; p.sC2 = sC2; p.zdiv = zdiv;
    p.ldcf = ldcf; p.ldcb = ldcb;
    p.bias = bias; p.sBias = sBias;
    p.gather = gather; p.gS = gS; p.cnt = cntp; p.gate = gate;
    p.M = M; p.N = N; p.K = K; p.alpha = alpha; p.flags = flags;
    dim3 g((unsigned)((N + 127) / 128), (unsigned)((M + 63) / 64), (unsigned)Z);
    gemm_f16_wmma<<<g, 256, 0, stream>>>(p);
  };

  for (int l = 0; l < Ln; ++l) {
    // --- attention ---
    rmsnorm_kernel<<<Tn, 256, 0, stream>>>(x, n1w + (size_t)l * Dn, hb);
    gemm(hb, 0, Dn, wqkvT + (size_t)l * 3 * Dn * Dn, 0, Dn, 3 * Dn,
         nullptr, qkvb, 0, 0, 1, 0, 3 * Dn,
         bqkv + (size_t)l * 3 * Dn, 0, nullptr, 0, nullptr, nullptr,
         Tn, 3 * Dn, Dn, 1.f, 0, 1);
    split_qkv_kernel<<<(Tn * Dn + 255) / 256, 256, 0, stream>>>(qkvb, qb, kb, vt);
    // scores = (Q K^T) * 1/sqrt(DH), causal-masked; per-(b,h) batch
    gemm(qb, (long long)Sq * DHn, DHn, kb, (long long)Sq * DHn, DHn, Sq,
         scF, nullptr, (long long)Sq * Sq, 0, 1, Sq, 0,
         nullptr, 0, nullptr, 0, nullptr, nullptr,
         Sq, Sq, DHn, 0.125f, WF_CAUSAL, Bq * Hn);
    softmax_rows_kernel<<<dim3(Sq, Bq * Hn), 256, 0, stream>>>(scF, pb);
    // o = P V, scattered straight into [B,S,D] f16 via two-level batch stride
    gemm(pb, (long long)Sq * Sq, Sq, vt, (long long)DHn * Sq, Sq, DHn,
         nullptr, ob, (long long)Sq * Dn, DHn, Hn, 0, Dn,
         nullptr, 0, nullptr, 0, nullptr, nullptr,
         Sq, DHn, Sq, 1.f, 0, Bq * Hn);
    // x += o Wo + bo
    gemm(ob, 0, Dn, woT + (size_t)l * Dn * Dn, 0, Dn, Dn,
         x, nullptr, 0, 0, 1, Dn, 0,
         bo + (size_t)l * Dn, 0, nullptr, 0, nullptr, nullptr,
         Tn, Dn, Dn, 1.f, WF_ACC, 1);

    // --- MoE ---
    rmsnorm_kernel<<<Tn, 256, 0, stream>>>(x, n2w + (size_t)l * Dn, hb);
    float* rlog_l = rlog_out + (size_t)l * Tn * En;
    gemm(hb, 0, Dn, rwT + (size_t)l * 16 * Dn, 0, Dn, 16,
         rlog_l, nullptr, 0, 0, 1, En, 0,
         rb + (size_t)l * En, 0, nullptr, 0, nullptr, nullptr,
         Tn, En, Dn, 1.f, 0, 1);
    zero32_kernel<<<1, 32, 0, stream>>>((unsigned*)cnt, 2 * En);
    router_top1_kernel<<<(Tn + 255) / 256, 256, 0, stream>>>(rlog_l, top_w, cnt,
                                                             lists, probSum);
    aux_accum_kernel<<<1, 32, 0, stream>>>(cnt, probSum, aux_out);
    // expert FFN on gathered tokens only (top-1 == reference dense+one-hot combine)
    gemm(hb, 0, Dn, w1T + (size_t)l * En * FFn * Dn, (long long)FFn * Dn, Dn, FFn,
         nullptr, hid, (long long)Tn * FFn, 0, 1, 0, FFn,
         eb1 + (size_t)l * En * FFn, FFn, lists, Tn, cnt, nullptr,
         Tn, FFn, Dn, 1.f, WF_GELU, En);
    gemm(hid, (long long)Tn * FFn, FFn, w2T + (size_t)l * En * Dn * FFn,
         (long long)Dn * FFn, FFn, Dn,
         x, nullptr, 0, 0, 1, Dn, 0,
         eb2 + (size_t)l * En * Dn, Dn, lists, Tn, cnt, top_w,
         Tn, Dn, FFn, 1.f, WF_ACC | WF_SCATTER, En);
  }

  // --- final norm + LM head ---
  rmsnorm_kernel<<<Tn, 256, 0, stream>>>(x, now, hb);
  gemm(hb, 0, Dn, lmWT, 0, Dn, Vn,
       out, nullptr, 0, 0, 1, Vn, 0,
       nullptr, 0, nullptr, 0, nullptr, nullptr,
       Tn, Vn, Dn, 1.f, 0, 1);
}
